// CausalGraphLearner_38182259262000
// MI455X (gfx1250) — matrix operand order repaired and checked by reading, not compiled, not run on hardware
//
#include <hip/hip_runtime.h>
#include <hip/hip_bf16.h>
#include <math.h>

typedef __attribute__((ext_vector_type(16))) _Float16 v16h;
typedef __attribute__((ext_vector_type(8)))  float    v8f;
typedef __attribute__((ext_vector_type(4)))  float    f4;

#define DIM 512
#define HID 1024
#define NB  64   // batch
#define NV  64   // vocab / graph nodes

// Workspace layout (floats):
//   [      0,  64*HID) cause_h
//   [ 64*HID, 128*HID) effect_h
//   [128*HID, 192*HID) ctx_h
//   [192*HID, 192*HID + 64*DIM) xsum = state + action
#define WS_XSUM (192 * HID)

// ---------------------------------------------------------------------------
// Kernel 0: xsum = state + action  (64 x 512)
// ---------------------------------------------------------------------------
__global__ __launch_bounds__(256)
void cgl_prep_kernel(const float* __restrict__ state,
                     const float* __restrict__ action,
                     float* __restrict__ xsum)
{
    const int idx = blockIdx.x * 256 + threadIdx.x;   // 0 .. 64*512-1
    xsum[idx] = state[idx] + action[idx];
}

// ---------------------------------------------------------------------------
// Kernel 1: three M=64 x K=512 x N=1024 GEMMs via v_wmma_f32_16x16x32_f16.
//   g=0: cause_h  = embed @ W1[   0: 512]
//   g=1: effect_h = embed @ W1[ 512:1024]
//   g=2: ctx_h    = xsum  @ W1[1024:1536]
// One wave computes one 16x16 output tile. 768 tiles total, 4 waves/block.
// A source is a wave-uniform pointer (scalar select, no divergence).
// ---------------------------------------------------------------------------
__global__ __launch_bounds__(128)
void cgl_gemm_kernel(const float* __restrict__ embed,
                     const float* __restrict__ W1,
                     float* __restrict__ ws)
{
    const int tile = blockIdx.x * 4 + (threadIdx.x >> 5);   // 0..767
    const int g    = tile / 256;                            // which GEMM
    const int t2   = tile % 256;
    const int m0   = (t2 / 64) * 16;                        // 0..48
    const int n0   = (t2 % 64) * 16;                        // 0..1008

    const int lane   = threadIdx.x & 31;
    const int laneLo = lane & 15;
    const int laneHi = (lane >> 4) << 3;   // K offset: 0 or 8

    const float* __restrict__ Wg   = W1 + (size_t)g * DIM * HID;  // [512][1024]
    const float* __restrict__ Asrc = (g < 2) ? embed : (ws + WS_XSUM);
    const float* __restrict__ Arow = Asrc + (size_t)(m0 + laneLo) * DIM;

    v8f acc = {};

    for (int k0 = 0; k0 < DIM; k0 += 32) {
        // --- A (16x32 f16): lane = row M; halves t=0..7 <- K=k0+laneHi+t,
        //     t=8..15 <- K=k0+16+laneHi+(t-8).  Two contiguous 8-float chunks.
        const f4* __restrict__ pa0 = (const f4*)(Arow + k0 + laneHi);
        const f4* __restrict__ pa1 = (const f4*)(Arow + k0 + 16 + laneHi);
        const f4 a0 = pa0[0], a1 = pa0[1], a2 = pa1[0], a3 = pa1[1];

        // --- B (32x16 f16): lane = col N=n0+laneLo; same K order as A.
        const float* __restrict__ pb = Wg + (size_t)(k0 + laneHi) * HID + n0 + laneLo;

        v16h a, b;
#pragma unroll
        for (int t = 0; t < 4; ++t) {
            a[t]      = (_Float16)a0[t];
            a[t + 4]  = (_Float16)a1[t];
            a[t + 8]  = (_Float16)a2[t];
            a[t + 12] = (_Float16)a3[t];
            b[t]      = (_Float16)pb[(size_t)t * HID];
            b[t + 4]  = (_Float16)pb[(size_t)(4 + t) * HID];
            b[t + 8]  = (_Float16)pb[(size_t)(16 + t) * HID];
            b[t + 12] = (_Float16)pb[(size_t)(20 + t) * HID];
        }

        acc = __builtin_amdgcn_wmma_f32_16x16x32_f16(
                  /*neg_a=*/false, a, /*neg_b=*/false, b,
                  /*c_mod=*/(short)0, acc, /*reuse_a=*/false, /*reuse_b=*/false);
    }

    // --- store C/D: VGPR r -> M = m0 + r + 8*(lane>=16); N = n0 + (lane&15)
    float* __restrict__ Out = ws + (size_t)g * NB * HID;
    const int rowBase = m0 + laneHi;   // laneHi is exactly the +8 row shift
#pragma unroll
    for (int r = 0; r < 8; ++r)
        Out[(size_t)(rowBase + r) * HID + n0 + laneLo] = acc[r];
}

// ---------------------------------------------------------------------------
// Kernel 2: fused scores[i,j] = mean_b sigmoid( b2 +
//              sum_h gelu(ctx[b,h] + cause[i,h] + effect[j,h] + b1[h]) * W2[h] )
// One block per (i,j); 256 threads = 4 threads per batch row b.
// ---------------------------------------------------------------------------
__global__ __launch_bounds__(256)
void cgl_score_kernel(const float* __restrict__ ws,
                      const float* __restrict__ b1,
                      const float* __restrict__ W2,
                      const float* __restrict__ b2,
                      float* __restrict__ out)
{
    __shared__ float s[HID];
    __shared__ float w2s[HID];
    __shared__ float partial[NB];

    const int i = blockIdx.x >> 6;   // cause index
    const int j = blockIdx.x & 63;   // effect index

    const float* __restrict__ cause  = ws + (size_t)i * HID;
    const float* __restrict__ effect = ws + (size_t)(NV + j) * HID;
    const float* __restrict__ ctx    = ws + (size_t)(2 * NV) * HID;

    for (int h = threadIdx.x; h < HID; h += 256) {
        s[h]   = cause[h] + effect[h] + b1[h];
        w2s[h] = W2[h];
    }
    __syncthreads();

    const int b = threadIdx.x >> 2;      // 0..63
    const int q = threadIdx.x & 3;       // 0..3
    const float* __restrict__ cb = ctx + (size_t)b * HID;

    const float inv_sqrt2 = 0.70710678118654752f;
    float acc = 0.0f;
#pragma unroll 4
    for (int h = q; h < HID; h += 4) {
        const float x = cb[h] + s[h];
        const float gx = 0.5f * x * (1.0f + erff(x * inv_sqrt2));  // exact gelu
        acc = fmaf(gx, w2s[h], acc);
    }

    // reduce the 4 partial sums per b (lanes b*4+q never straddle a wave32)
    acc += __shfl_xor(acc, 1, 32);
    acc += __shfl_xor(acc, 2, 32);

    if (q == 0) {
        const float logit = acc + b2[0];
        partial[b] = 1.0f / (1.0f + expf(-logit));
    }
    __syncthreads();

    if (threadIdx.x == 0) {
        float sum = 0.0f;
#pragma unroll
        for (int k = 0; k < NB; ++k) sum += partial[k];
        out[blockIdx.x] = sum * (1.0f / (float)NB);
    }
}

// ---------------------------------------------------------------------------
// Launch. Inputs (setup_inputs order):
//   0: state (64,512) f32   1: action (64,512) f32   2: embed (64,512) f32
//   3: W1 (1536,1024) f32   4: b1 (1024) f32         5: W2 (1024,1) f32
//   6: b2 (1) f32
// Output: scores (64,64) f32.
// ---------------------------------------------------------------------------
extern "C" void kernel_launch(void* const* d_in, const int* in_sizes, int n_in,
                              void* d_out, int out_size, void* d_ws, size_t ws_size,
                              hipStream_t stream)
{
    const float* state  = (const float*)d_in[0];
    const float* action = (const float*)d_in[1];
    const float* embed  = (const float*)d_in[2];
    const float* W1     = (const float*)d_in[3];
    const float* b1     = (const float*)d_in[4];
    const float* W2     = (const float*)d_in[5];
    const float* b2     = (const float*)d_in[6];
    float*       out    = (float*)d_out;
    float*       ws     = (float*)d_ws;

    // xsum = state + action (64x512 = 32768 elements)
    cgl_prep_kernel<<<(NB * DIM) / 256, 256, 0, stream>>>(state, action, ws + WS_XSUM);

    // 768 WMMA tiles, 4 waves (one tile each) per 128-thread block -> 192 blocks
    cgl_gemm_kernel<<<192, 128, 0, stream>>>(embed, W1, ws);

    // one block per (i,j) score
    cgl_score_kernel<<<NV * NV, 256, 0, stream>>>(ws, b1, W2, b2, out);
}
